// SIREConv_2645699854682
// MI455X (gfx1250) — matrix operand order repaired and checked by reading, not compiled, not run on hardware
//
#include <hip/hip_runtime.h>
#include <hip/hip_bf16.h>

// ---------------------------------------------------------------------------
// SIREConv on MI455X (gfx1250): bf16 WMMA GEMMs + fused gather/ReLU/scatter
//   round 3: explicit double-buffering of B fragments across K-steps so
//   v_wmma consumes loads issued a full iteration earlier (no load->wait->wmma
//   serialization on L2 latency).
// ---------------------------------------------------------------------------

typedef __bf16 bf16_t;
typedef __attribute__((ext_vector_type(16))) __bf16 v16bf;
typedef __attribute__((ext_vector_type(8)))  __bf16 v8bf;
typedef __attribute__((ext_vector_type(4)))  __bf16 v4bf;
typedef __attribute__((ext_vector_type(8)))  float  v8f;

#define DIM      256
#define NNODES   10000
#define NEDGES   160000
#define NODE_TILES (NNODES / 16)     // 625
#define EDGE_BLK   64                // edges per block in edge kernel
#define EDGE_BLKS  (NEDGES / EDGE_BLK)  // 2500

// ---- WMMA helper: D = A(16x32 bf16) * B(32x16 bf16) + C(16x16 f32) --------
static __device__ __forceinline__ v8f wmma_bf16(v16bf a, v16bf b, v8f c) {
  return __builtin_amdgcn_wmma_f32_16x16x32_bf16(
      /*neg_a=*/false, a, /*neg_b=*/false, b,
      /*c_mod=*/(short)0, c, /*reuse_a=*/false, /*reuse_b=*/false);
}

// ---- A fragment from LDS (rows x DIM bf16, row-major); row0 = tile origin --
// ISA 16-bit A 16x32 layout: lane<16 row M=lane holds K=klo+[0..7] then
// K=klo+[16..23]; lane>=16 holds K=klo+[8..15] then K=klo+[24..31].
static __device__ __forceinline__ v16bf load_a_frag(const bf16_t* lds, int row0,
                                                    int lane, int klo) {
  const int row  = row0 + (lane & 15);
  const int base = row * DIM + klo + ((lane >> 4) << 3);
  v8bf lo = *(const v8bf*)(lds + base);        // ds_load_b128
  v8bf hi = *(const v8bf*)(lds + base + 16);   // ds_load_b128
  return __builtin_shufflevector(lo, hi, 0,1,2,3,4,5,6,7,8,9,10,11,12,13,14,15);
}

// ---- B fragment from bf16 column-major weights (Wp[n*DIM + k]) ------------
// ISA 16-bit B 32x16 layout: lane gives N=lane&15; lane<16 holds K=klo+[0..7] then
// K=klo+[16..23]; lane>=16 holds K=klo+[8..15] then K=klo+[24..31].
static __device__ __forceinline__ v16bf load_b_frag(const bf16_t* __restrict__ Wp,
                                                    int lane, int klo, int n0) {
  const int n    = n0 + (lane & 15);
  const int base = n * DIM + klo + ((lane >> 4) << 3);
  v8bf lo = *(const v8bf*)(Wp + base);         // global_load_b128
  v8bf hi = *(const v8bf*)(Wp + base + 16);    // global_load_b128
  return __builtin_shufflevector(lo, hi, 0,1,2,3,4,5,6,7,8,9,10,11,12,13,14,15);
}

// ---- block-wide stage: NROWS x DIM f32 (contiguous) -> bf16 LDS -----------
template <int NROWS>
static __device__ __forceinline__ void stage_rows_f32_to_bf16(
    const float* __restrict__ gsrc, bf16_t* lds, int tid) {
  const float4* g4 = (const float4*)gsrc;
  #pragma unroll
  for (int i = tid; i < (NROWS * DIM) / 4; i += 256) {
    float4 v = g4[i];                           // global_load_b128
    v4bf b;
    b[0] = (bf16_t)v.x; b[1] = (bf16_t)v.y;
    b[2] = (bf16_t)v.z; b[3] = (bf16_t)v.w;
    ((v4bf*)lds)[i] = b;                        // ds_store_b64
  }
}

static __device__ __forceinline__ v8f splat8(float x) {
  v8f a;
  #pragma unroll
  for (int r = 0; r < 8; ++r) a[r] = x;
  return a;
}

// ---------------------------------------------------------------------------
// Kernel 0: convert one 256x256 f32 row-major weight -> bf16 column-major
// ---------------------------------------------------------------------------
__global__ void __launch_bounds__(256)
k_prep_w(const float* __restrict__ W, bf16_t* __restrict__ out) {
  const int idx = blockIdx.x * 256 + threadIdx.x;  // 65536 elements
  const int n = idx >> 8;
  const int k = idx & 255;
  out[n * DIM + k] = (bf16_t)W[k * DIM + n];
}

// ---------------------------------------------------------------------------
// Kernel 1: zero the segment-sum accumulator
// ---------------------------------------------------------------------------
__global__ void __launch_bounds__(256)
k_zero(float4* __restrict__ p, int n4) {
  const int i = blockIdx.x * 256 + threadIdx.x;
  if (i < n4) p[i] = make_float4(0.f, 0.f, 0.f, 0.f);
}

// ---------------------------------------------------------------------------
// Kernel 2: eq = nfeat@Wq + bq ; ek = nfeat@Wk + bk   (both [NNODES, 256])
// grid.x = node tile (16 nodes); waves 0-3 -> q, 4-7 -> k; 4 N-tiles each.
// B fragments double-buffered across K-steps.
// ---------------------------------------------------------------------------
__global__ void __launch_bounds__(256)
k_node_qk(const float* __restrict__ nfeat,
          const bf16_t* __restrict__ WqT, const bf16_t* __restrict__ WkT,
          const float* __restrict__ bq,   const float* __restrict__ bk,
          float* __restrict__ eq,         float* __restrict__ ek) {
  __shared__ bf16_t ldsA[16 * DIM];
  const int tid   = threadIdx.x;
  const int node0 = blockIdx.x * 16;

  stage_rows_f32_to_bf16<16>(nfeat + (size_t)node0 * DIM, ldsA, tid);
  __syncthreads();

  const int wave = tid >> 5;
  const int lane = tid & 31;
  const bool is_q = (wave < 4);
  const bf16_t* Wp   = is_q ? WqT : WkT;
  const float*  bias = is_q ? bq  : bk;
  float*        outp = is_q ? eq  : ek;
  const int nbase = (wave & 3) * 64;          // 4 tiles of 16 columns
  const int n     = lane & 15;
  const int half  = lane >> 4;

  v8f acc[4];
  #pragma unroll
  for (int j = 0; j < 4; ++j) acc[j] = splat8(bias[nbase + j * 16 + n]);

  // prologue: B fragments for klo = 0
  v16bf b[4];
  #pragma unroll
  for (int j = 0; j < 4; ++j) b[j] = load_b_frag(Wp, lane, 0, nbase + j * 16);

  #pragma unroll
  for (int klo = 0; klo < DIM; klo += 32) {
    v16bf cb[4];
    #pragma unroll
    for (int j = 0; j < 4; ++j) cb[j] = b[j];
    if (klo + 32 < DIM) {
      #pragma unroll
      for (int j = 0; j < 4; ++j)
        b[j] = load_b_frag(Wp, lane, klo + 32, nbase + j * 16);
    }
    v16bf a = load_a_frag(ldsA, 0, lane, klo);
    #pragma unroll
    for (int j = 0; j < 4; ++j) acc[j] = wmma_bf16(a, cb[j], acc[j]);
  }

  #pragma unroll
  for (int j = 0; j < 4; ++j) {
    const int nc = nbase + j * 16 + n;
    #pragma unroll
    for (int r = 0; r < 8; ++r) {
      const int m = r + 8 * half;
      outp[(size_t)(node0 + m) * DIM + nc] = acc[j][r];
    }
  }
}

// ---------------------------------------------------------------------------
// Kernel 3: per-edge m = relu(efeat@We + be + eq[dst] + ek[src]); ft += scatter
// grid.x = 64-edge block; 8 waves, each: 2 N-tiles x 4 M-tiles
//   -> each B fragment pair feeds 8 WMMAs; B double-buffered across K-steps.
// ---------------------------------------------------------------------------
__global__ void __launch_bounds__(256)
k_edge(const float* __restrict__ efeat, const bf16_t* __restrict__ WeT,
       const float* __restrict__ be,
       const int* __restrict__ src, const int* __restrict__ dst,
       const float* __restrict__ eq, const float* __restrict__ ek,
       float* __restrict__ ft) {
  __shared__ bf16_t ldsA[EDGE_BLK * DIM];      // 32 KB
  __shared__ int sSrc[EDGE_BLK];
  __shared__ int sDst[EDGE_BLK];
  const int tid = threadIdx.x;
  const int e0  = blockIdx.x * EDGE_BLK;

  // pull the next 64-edge tile (64 KB) toward L2 while we work on this one
  if (blockIdx.x + 1 < EDGE_BLKS) {
    const float* nxt = efeat + (size_t)(e0 + EDGE_BLK) * DIM;
    #pragma unroll
    for (int i = 0; i < 4; ++i)
      __builtin_prefetch(nxt + (size_t)(tid * 4 + i) * 16, 0, 1);
  }

  stage_rows_f32_to_bf16<EDGE_BLK>(efeat + (size_t)e0 * DIM, ldsA, tid);
  if (tid < EDGE_BLK) {
    sSrc[tid] = src[e0 + tid];
    sDst[tid] = dst[e0 + tid];
  }
  __syncthreads();

  const int wave = tid >> 5;
  const int lane = tid & 31;
  const int n    = lane & 15;
  const int half = lane >> 4;
  const int n0   = wave * 32;

  v8f acc[4][2];                                // [M-tile][N-tile]
  #pragma unroll
  for (int j = 0; j < 2; ++j) {
    const float bv = be[n0 + j * 16 + n];
    #pragma unroll
    for (int mi = 0; mi < 4; ++mi) acc[mi][j] = splat8(bv);
  }

  // prologue: B fragments for klo = 0
  v16bf b0 = load_b_frag(WeT, lane, 0, n0);
  v16bf b1 = load_b_frag(WeT, lane, 0, n0 + 16);

  #pragma unroll
  for (int klo = 0; klo < DIM; klo += 32) {
    v16bf cb0 = b0, cb1 = b1;
    if (klo + 32 < DIM) {
      b0 = load_b_frag(WeT, lane, klo + 32, n0);
      b1 = load_b_frag(WeT, lane, klo + 32, n0 + 16);
    }
    #pragma unroll
    for (int mi = 0; mi < 4; ++mi) {
      v16bf a = load_a_frag(ldsA, mi * 16, lane, klo);
      acc[mi][0] = wmma_bf16(a, cb0, acc[mi][0]);
      acc[mi][1] = wmma_bf16(a, cb1, acc[mi][1]);
    }
  }

  // epilogue: gather dst/src node embeddings, relu, scatter-sum (L2-resident)
  #pragma unroll
  for (int mi = 0; mi < 4; ++mi) {
    #pragma unroll
    for (int j = 0; j < 2; ++j) {
      const int nc = n0 + j * 16 + n;
      #pragma unroll
      for (int r = 0; r < 8; ++r) {
        const int m = mi * 16 + r + 8 * half;   // edge within 64-edge block
        const int d = sDst[m];
        const int s = sSrc[m];
        float v = acc[mi][j][r] + eq[(size_t)d * DIM + nc] + ek[(size_t)s * DIM + nc];
        v = fmaxf(v, 0.0f);
        (void)__hip_atomic_fetch_add(&ft[(size_t)d * DIM + nc], v,
                                     __ATOMIC_RELAXED, __HIP_MEMORY_SCOPE_AGENT);
      }
    }
  }
}

// ---------------------------------------------------------------------------
// Kernel 4: out = ft@Wr + br   [NNODES, 256]; B double-buffered
// ---------------------------------------------------------------------------
__global__ void __launch_bounds__(256)
k_out(const float* __restrict__ ft, const bf16_t* __restrict__ WrT,
      const float* __restrict__ br, float* __restrict__ out) {
  __shared__ bf16_t ldsA[16 * DIM];
  const int tid   = threadIdx.x;
  const int node0 = blockIdx.x * 16;

  stage_rows_f32_to_bf16<16>(ft + (size_t)node0 * DIM, ldsA, tid);
  __syncthreads();

  const int wave = tid >> 5;
  const int lane = tid & 31;
  const int n    = lane & 15;
  const int half = lane >> 4;
  const int n0   = wave * 32;

  v8f acc[2];
  #pragma unroll
  for (int j = 0; j < 2; ++j) acc[j] = splat8(br[n0 + j * 16 + n]);

  v16bf b0 = load_b_frag(WrT, lane, 0, n0);
  v16bf b1 = load_b_frag(WrT, lane, 0, n0 + 16);

  #pragma unroll
  for (int klo = 0; klo < DIM; klo += 32) {
    v16bf cb0 = b0, cb1 = b1;
    if (klo + 32 < DIM) {
      b0 = load_b_frag(WrT, lane, klo + 32, n0);
      b1 = load_b_frag(WrT, lane, klo + 32, n0 + 16);
    }
    v16bf a = load_a_frag(ldsA, 0, lane, klo);
    acc[0] = wmma_bf16(a, cb0, acc[0]);
    acc[1] = wmma_bf16(a, cb1, acc[1]);
  }

  #pragma unroll
  for (int j = 0; j < 2; ++j) {
    const int nc = n0 + j * 16 + n;
    #pragma unroll
    for (int r = 0; r < 8; ++r) {
      const int m = r + 8 * half;
      out[(size_t)(node0 + m) * DIM + nc] = acc[j][r];
    }
  }
}

// ---------------------------------------------------------------------------
// Host launcher
// ---------------------------------------------------------------------------
extern "C" void kernel_launch(void* const* d_in, const int* in_sizes, int n_in,
                              void* d_out, int out_size, void* d_ws, size_t ws_size,
                              hipStream_t stream) {
  const float* nfeat = (const float*)d_in[0];
  const float* efeat = (const float*)d_in[1];
  const int*   src   = (const int*)  d_in[2];
  const int*   dst   = (const int*)  d_in[3];
  const float* Wq    = (const float*)d_in[4];
  const float* bq    = (const float*)d_in[5];
  const float* Wk    = (const float*)d_in[6];
  const float* bk    = (const float*)d_in[7];
  const float* We    = (const float*)d_in[8];
  const float* be    = (const float*)d_in[9];
  const float* Wr    = (const float*)d_in[10];
  const float* br    = (const float*)d_in[11];

  // workspace layout:
  //   4 x bf16 col-major weights (128 KB each) | eq | ek | ft (f32, 10.24 MB each)
  char* ws = (char*)d_ws;
  bf16_t* WqT = (bf16_t*)ws;
  bf16_t* WkT = WqT + DIM * DIM;
  bf16_t* WeT = WkT + DIM * DIM;
  bf16_t* WrT = WeT + DIM * DIM;
  float* eq = (float*)(ws + 4 * (size_t)DIM * DIM * sizeof(bf16_t));
  float* ek = eq + (size_t)NNODES * DIM;
  float* ft = ek + (size_t)NNODES * DIM;

  k_prep_w<<<256, 256, 0, stream>>>(Wq, WqT);
  k_prep_w<<<256, 256, 0, stream>>>(Wk, WkT);
  k_prep_w<<<256, 256, 0, stream>>>(We, WeT);
  k_prep_w<<<256, 256, 0, stream>>>(Wr, WrT);

  k_zero<<<(NNODES * DIM / 4 + 255) / 256, 256, 0, stream>>>((float4*)ft, NNODES * DIM / 4);

  k_node_qk<<<NODE_TILES, 256, 0, stream>>>(nfeat, WqT, WkT, bq, bk, eq, ek);
  k_edge<<<EDGE_BLKS, 256, 0, stream>>>(efeat, WeT, be, src, dst, eq, ek, ft);
  k_out<<<NODE_TILES, 256, 0, stream>>>(ft, WrT, br, (float*)d_out);
}
